// MultiHeadTargetCentricAttention_18691697672859
// MI455X (gfx1250) — matrix operand order repaired and checked by reading, not compiled, not run on hardware
//
#include <hip/hip_runtime.h>
#include <hip/hip_bf16.h>

// ---------------- types ----------------
typedef __bf16 bf16_t;
typedef __attribute__((ext_vector_type(16))) __bf16 v16bf;
typedef __attribute__((ext_vector_type(8)))  float  v8f;

union frag16 {
    v16bf  v;
    uint4  q[2];
    bf16_t e[16];
};

union pack8 {
    uint4  u;
    bf16_t e[8];
};

__device__ __forceinline__ bf16_t f2bf(float f) {
    unsigned u = __builtin_bit_cast(unsigned, f);
    u += 0x7fffu + ((u >> 16) & 1u);                 // round-to-nearest-even
    unsigned short h = (unsigned short)(u >> 16);
    return __builtin_bit_cast(bf16_t, h);
}

__device__ __forceinline__ v8f wmma_bf16(const frag16& a, const frag16& b, v8f c) {
    return __builtin_amdgcn_wmma_f32_16x16x32_bf16(
        /*neg_a=*/false, a.v, /*neg_b=*/false, b.v,
        /*c_mod=*/(short)0, c, /*reuse_a=*/false, /*reuse_b=*/false);
}

// CDNA5 async global -> LDS copy (16B per lane), tracked by ASYNCcnt.
__device__ __forceinline__ void async_b128(unsigned lds_off, const void* g) {
    unsigned long long ga = (unsigned long long)g;
    asm volatile("global_load_async_to_lds_b128 %0, %1, off"
                 :: "v"(lds_off), "v"(ga) : "memory");
}
__device__ __forceinline__ void wait_async0() {
    asm volatile("s_wait_asynccnt 0x0" ::: "memory");
}

// ---------------- problem constants ----------------
#define BATCH   4
#define SEQ     2048
#define DMODEL  1024
#define NHEADS  16
#define DHEAD   64
#define BS      (BATCH * SEQ)          // 8192

// ---------------- fp32 -> bf16 convert ----------------
__global__ void cvt_f32_bf16(const float* __restrict__ src,
                             bf16_t* __restrict__ dst, int n) {
    int i = blockIdx.x * blockDim.x + threadIdx.x;
    int stride = gridDim.x * blockDim.x;
    for (; i < n; i += stride) dst[i] = f2bf(src[i]);
}

// fp32 W[R][C] -> bf16 W^T[C][R]  (writes coalesced; one-off 4MB per weight)
__global__ void cvtT_f32_bf16(const float* __restrict__ src,
                              bf16_t* __restrict__ dst, int R, int C) {
    int i = blockIdx.x * blockDim.x + threadIdx.x;
    int stride = gridDim.x * blockDim.x;
    int n = R * C;
    for (; i < n; i += stride) {
        int c = i / R, r = i - c * R;              // dst[c][r] = src[r][c]
        dst[i] = f2bf(src[(size_t)r * C + c]);
    }
}

// ---------------- bf16 GEMM: C[M,N] = A[M,K] @ Bt[N,K]^T + bias -------------
// block tile 128x128, K-step 32, 256 threads = 8 waves (2x4 wave grid),
// each wave 64x32 = 4x2 wmma tiles. Double-buffered async global->LDS staging.
#define TM 128
#define TN 128
#define TK 32
#define LDS_BUF_BYTES 16384          // 8KB A + 8KB B per buffer

__global__ void __launch_bounds__(256)
gemm_bf16(const bf16_t* __restrict__ A, const bf16_t* __restrict__ Bt,
          const float* __restrict__ bias,
          float* __restrict__ Cf, bf16_t* __restrict__ Cb,
          bf16_t* __restrict__ CbT,          // transposed [b,h,dh,s] bf16 output
          int M, int N, int K) {
    extern __shared__ char smem[];   // 2 * 16KB (dynamic => LDS offset 0)

    const int tid  = threadIdx.x;
    const int lane = tid & 31;
    const int wave = tid >> 5;
    const int half = lane >> 4;
    const int l15  = lane & 15;

    const int bm = blockIdx.x * TM;
    const int bn = blockIdx.y * TN;
    const int wm = (wave >> 2) * 64;   // 0 or 64
    const int wn = (wave & 3) * 32;    // 0,32,64,96

    // async staging: thread t -> row t/2 of both tiles, 2x16B chunks
    const int srow = tid >> 1, scol = (tid & 1) * 16;

    auto stage = [&](int kt, int db) {
        unsigned aoff = (unsigned)(db * LDS_BUF_BYTES + srow * 64 + scol * 2);
        const bf16_t* ga = &A[(size_t)(bm + srow) * K + kt + scol];
        async_b128(aoff,      ga);
        async_b128(aoff + 16, ga + 8);
        unsigned boff = aoff + 8192u;
        const bf16_t* gb = &Bt[(size_t)(bn + srow) * K + kt + scol];
        async_b128(boff,      gb);
        async_b128(boff + 16, gb + 8);
    };

    v8f acc[4][2] = {};

    stage(0, 0);
    wait_async0();
    __syncthreads();

    const int nk = K / TK;
    for (int i = 0; i < nk; ++i) {
        const int db = i & 1;
        if (i + 1 < nk) stage((i + 1) * TK, db ^ 1);   // overlap with compute

        const bf16_t* lAp = reinterpret_cast<const bf16_t*>(smem + db * LDS_BUF_BYTES);
        const bf16_t* lBp = lAp + 4096;                // +8192 bytes

        frag16 af[4];
#pragma unroll
        for (int mt = 0; mt < 4; ++mt) {
            const bf16_t* p = &lAp[(wm + mt * 16 + l15) * 32];
            af[mt].q[0] = *reinterpret_cast<const uint4*>(p + half * 16);
            af[mt].q[1] = *reinterpret_cast<const uint4*>(p + half * 16 + 8);
        }
        frag16 bf[2];
#pragma unroll
        for (int nt = 0; nt < 2; ++nt) {
            const bf16_t* p = &lBp[(wn + nt * 16 + l15) * 32 + half * 16];
            bf[nt].q[0] = *reinterpret_cast<const uint4*>(p);
            bf[nt].q[1] = *reinterpret_cast<const uint4*>(p + 8);
        }
#pragma unroll
        for (int mt = 0; mt < 4; ++mt)
#pragma unroll
            for (int nt = 0; nt < 2; ++nt)
                acc[mt][nt] = wmma_bf16(af[mt], bf[nt], acc[mt][nt]);

        if (i + 1 < nk) wait_async0();   // next tile resident before barrier
        __syncthreads();
    }

    // ---- epilogue: vgpr r holds row (r + half*8), col = l15
#pragma unroll
    for (int mt = 0; mt < 4; ++mt)
#pragma unroll
        for (int nt = 0; nt < 2; ++nt) {
            int col = bn + wn + nt * 16 + l15;
            float bv = bias[col];
            if (CbT) {
                // V path: write V^T[b, h, dh, s]; 8 consecutive s => 16B store
                int row0 = bm + wm + mt * 16 + half * 8;
                int b    = row0 / SEQ;
                int s0   = row0 - b * SEQ;
                int h    = col >> 6, j = col & 63;
                pack8 pk;
#pragma unroll
                for (int r = 0; r < 8; ++r) pk.e[r] = f2bf(acc[mt][nt][r] + bv);
                *reinterpret_cast<uint4*>(
                    &CbT[(((size_t)b * NHEADS + h) * DHEAD + j) * SEQ + s0]) = pk.u;
            } else {
#pragma unroll
                for (int r = 0; r < 8; ++r) {
                    int row = bm + wm + mt * 16 + r + half * 8;
                    float val = acc[mt][nt][r] + bv;
                    if (Cb) Cb[(size_t)row * N + col] = f2bf(val);
                    else    Cf[(size_t)row * N + col] = val;
                }
            }
        }
}

// ---------------- attention pass 1: per-q softmax stats over k --------------
__global__ void __launch_bounds__(256)
attn_stats(const bf16_t* __restrict__ qm, const bf16_t* __restrict__ km,
           float* __restrict__ m_out, float* __restrict__ z_out) {
    const int bh = blockIdx.x;
    const int b  = bh >> 4, h = bh & 15;
    const int wave = threadIdx.x >> 5, lane = threadIdx.x & 31;
    const int half = lane >> 4, l15 = lane & 15;
    const int q0 = (blockIdx.y * 8 + wave) * 16;

    const bf16_t* qbase = qm + (size_t)b * SEQ * DMODEL + (size_t)h * DHEAD;
    const bf16_t* kbase = km + (size_t)b * SEQ * DMODEL + (size_t)h * DHEAD;

    frag16 aq[2];
#pragma unroll
    for (int kk = 0; kk < 2; ++kk) {
        const bf16_t* p = qbase + (size_t)(q0 + l15) * DMODEL + kk * 32;
        aq[kk].q[0] = *reinterpret_cast<const uint4*>(p + half * 16);
        aq[kk].q[1] = *reinterpret_cast<const uint4*>(p + half * 16 + 8);
    }

    const float scale = 0.125f;   // 1/sqrt(64)
    float m[8], z[8];
#pragma unroll
    for (int r = 0; r < 8; ++r) { m[r] = -1e30f; z[r] = 0.0f; }

    for (int k0 = 0; k0 < SEQ; k0 += 16) {
        frag16 bk[2];
#pragma unroll
        for (int kk = 0; kk < 2; ++kk) {
            const bf16_t* p = kbase + (size_t)(k0 + l15) * DMODEL + kk * 32 + half * 16;
            bk[kk].q[0] = *reinterpret_cast<const uint4*>(p);
            bk[kk].q[1] = *reinterpret_cast<const uint4*>(p + 8);
        }
        v8f s = {};
        s = wmma_bf16(aq[0], bk[0], s);
        s = wmma_bf16(aq[1], bk[1], s);

#pragma unroll
        for (int r = 0; r < 8; ++r) {
            float sv = s[r] * scale;
            float rm = sv;
#pragma unroll
            for (int msk = 1; msk <= 8; msk <<= 1)
                rm = fmaxf(rm, __shfl_xor(rm, msk, 32));
            float nm = fmaxf(m[r], rm);
            float ev = __expf(sv - nm);
            float rs = ev;
#pragma unroll
            for (int msk = 1; msk <= 8; msk <<= 1)
                rs += __shfl_xor(rs, msk, 32);
            z[r] = z[r] * __expf(m[r] - nm) + rs;
            m[r] = nm;
        }
    }
    if (l15 == 0) {
#pragma unroll
        for (int r = 0; r < 8; ++r) {
            int qi = q0 + r + half * 8;
            m_out[(size_t)bh * SEQ + qi] = m[r];
            z_out[(size_t)bh * SEQ + qi] = z[r];
        }
    }
}

// ---------------- attention pass 2: out[k,d] = sum_q P[q,k] * V[q,d] --------
// S-tile C-layout == P^T A-layout per lane -> no cross-lane shuffles.
// V supplied transposed: vt[b, h, dh, s] -> B fragments are contiguous b128s.
__global__ void __launch_bounds__(256)
attn_out(const bf16_t* __restrict__ qm, const bf16_t* __restrict__ km,
         const bf16_t* __restrict__ vt,
         const float* __restrict__ m_in, const float* __restrict__ z_in,
         bf16_t* __restrict__ out) {
    const int bh = blockIdx.x;
    const int b  = bh >> 4, h = bh & 15;
    const int wave = threadIdx.x >> 5, lane = threadIdx.x & 31;
    const int half = lane >> 4, l15 = lane & 15;
    const int k0 = (blockIdx.y * 8 + wave) * 16;

    const bf16_t* qbase  = qm + (size_t)b * SEQ * DMODEL + (size_t)h * DHEAD;
    const bf16_t* kbase  = km + (size_t)b * SEQ * DMODEL + (size_t)h * DHEAD;
    const bf16_t* vtbase = vt + ((size_t)b * NHEADS + h) * DHEAD * SEQ;
    const float*  mrow   = m_in + (size_t)bh * SEQ;
    const float*  zrow   = z_in + (size_t)bh * SEQ;

    frag16 bs[2];
#pragma unroll
    for (int kk = 0; kk < 2; ++kk) {
        const bf16_t* p = kbase + (size_t)(k0 + l15) * DMODEL + kk * 32 + half * 16;
        bs[kk].q[0] = *reinterpret_cast<const uint4*>(p);
        bs[kk].q[1] = *reinterpret_cast<const uint4*>(p + 8);
    }

    const float scale = 0.125f;
    v8f acc[4] = {};

    for (int qc = 0; qc < SEQ; qc += 32) {
        v8f s2[2];
#pragma unroll
        for (int part = 0; part < 2; ++part) {
            frag16 a0, a1;
            const bf16_t* p0 = qbase + (size_t)(qc + part * 16 + l15) * DMODEL;
            a0.q[0] = *reinterpret_cast<const uint4*>(p0 + half * 16);
            a0.q[1] = *reinterpret_cast<const uint4*>(p0 + 8 + half * 16);
            a1.q[0] = *reinterpret_cast<const uint4*>(p0 + 32 + half * 16);
            a1.q[1] = *reinterpret_cast<const uint4*>(p0 + 40 + half * 16);
            v8f s = {};
            s = wmma_bf16(a0, bs[0], s);
            s = wmma_bf16(a1, bs[1], s);
            s2[part] = s;
        }
        frag16 a2;
#pragma unroll
        for (int r = 0; r < 8; ++r) {
            int qlo = qc + r + half * 8;
            int qhi = qlo + 16;
            float plo = __expf(s2[0][r] * scale - mrow[qlo]) / zrow[qlo];
            float phi = __expf(s2[1][r] * scale - mrow[qhi]) / zrow[qhi];
            a2.e[r]     = f2bf(plo);
            a2.e[8 + r] = f2bf(phi);
        }
        int qr0 = qc + half * 8;
#pragma unroll
        for (int n = 0; n < 4; ++n) {
            frag16 b2;
            const bf16_t* p = vtbase + (size_t)(n * 16 + l15) * SEQ + qr0;
            b2.q[0] = *reinterpret_cast<const uint4*>(p);
            b2.q[1] = *reinterpret_cast<const uint4*>(p + 16);
            acc[n] = wmma_bf16(a2, b2, acc[n]);
        }
    }

#pragma unroll
    for (int n = 0; n < 4; ++n)
#pragma unroll
        for (int r = 0; r < 8; ++r) {
            int krow = k0 + r + half * 8;
            out[((size_t)b * SEQ + krow) * DMODEL + h * DHEAD + n * 16 + l15] =
                f2bf(acc[n][r]);
        }
}

// ---------------- host side ----------------
extern "C" void kernel_launch(void* const* d_in, const int* in_sizes, int n_in,
                              void* d_out, int out_size, void* d_ws, size_t ws_size,
                              hipStream_t stream) {
    const float* x  = (const float*)d_in[0];
    const float* Wq = (const float*)d_in[1];
    const float* bq = (const float*)d_in[2];
    const float* Wk = (const float*)d_in[3];
    const float* bk = (const float*)d_in[4];
    const float* Wv = (const float*)d_in[5];
    const float* bv = (const float*)d_in[6];
    const float* Wo = (const float*)d_in[7];
    const float* bo = (const float*)d_in[8];
    float* out = (float*)d_out;

    char* p = (char*)d_ws;
    auto carve = [&](size_t bytes) -> void* {
        void* r = (void*)p;
        p += (bytes + 255) & ~(size_t)255;
        return r;
    };
    bf16_t* xb   = (bf16_t*)carve((size_t)BS * DMODEL * 2);
    bf16_t* wqt  = (bf16_t*)carve((size_t)DMODEL * DMODEL * 2);
    bf16_t* wkt  = (bf16_t*)carve((size_t)DMODEL * DMODEL * 2);
    bf16_t* wvt  = (bf16_t*)carve((size_t)DMODEL * DMODEL * 2);
    bf16_t* wot  = (bf16_t*)carve((size_t)DMODEL * DMODEL * 2);
    bf16_t* qb   = (bf16_t*)carve((size_t)BS * DMODEL * 2);
    bf16_t* kb   = (bf16_t*)carve((size_t)BS * DMODEL * 2);
    bf16_t* vt   = (bf16_t*)carve((size_t)BS * DMODEL * 2);   // [b,h,dh,s]
    float*  mbuf = (float*) carve((size_t)BATCH * NHEADS * SEQ * 4);
    float*  zbuf = (float*) carve((size_t)BATCH * NHEADS * SEQ * 4);
    bf16_t* atb  = (bf16_t*)carve((size_t)BS * DMODEL * 2);

    // 1) fp32 -> bf16 (x plain; weights transposed to [N][K])
    int nx = BS * DMODEL, nw = DMODEL * DMODEL;
    cvt_f32_bf16 <<<1024, 256, 0, stream>>>(x,  xb,  nx);
    cvtT_f32_bf16<<<512,  256, 0, stream>>>(Wq, wqt, DMODEL, DMODEL);
    cvtT_f32_bf16<<<512,  256, 0, stream>>>(Wk, wkt, DMODEL, DMODEL);
    cvtT_f32_bf16<<<512,  256, 0, stream>>>(Wv, wvt, DMODEL, DMODEL);
    cvtT_f32_bf16<<<512,  256, 0, stream>>>(Wo, wot, DMODEL, DMODEL);

    // 2) Q/K projections (bf16 row-major), V projection (bf16 transposed)
    dim3 gg(BS / TM, DMODEL / TN);
    size_t shmem = 2 * LDS_BUF_BYTES;
    gemm_bf16<<<gg, 256, shmem, stream>>>(xb, wqt, bq, nullptr, qb, nullptr,
                                          BS, DMODEL, DMODEL);
    gemm_bf16<<<gg, 256, shmem, stream>>>(xb, wkt, bk, nullptr, kb, nullptr,
                                          BS, DMODEL, DMODEL);
    gemm_bf16<<<gg, 256, shmem, stream>>>(xb, wvt, bv, nullptr, nullptr, vt,
                                          BS, DMODEL, DMODEL);

    // 3) attention: softmax stats, then target-centric aggregation
    dim3 ga(BATCH * NHEADS, SEQ / (16 * 8));
    attn_stats<<<ga, 256, 0, stream>>>(qb, kb, mbuf, zbuf);
    attn_out  <<<ga, 256, 0, stream>>>(qb, kb, vt, mbuf, zbuf, atb);

    // 4) output projection (fp32 -> d_out)
    gemm_bf16<<<gg, 256, shmem, stream>>>(atb, wot, bo, out, nullptr, nullptr,
                                          BS, DMODEL, DMODEL);
}
